// TransformerConvStyleLayer_58634893525425
// MI455X (gfx1250) — compile-verified, matrix operand order checked by reading, not run on hardware
//
#include <hip/hip_runtime.h>
#include <hip/hip_bf16.h>

#define DFEAT 128
#define LN_EPS 1e-5f

typedef __attribute__((ext_vector_type(16))) __bf16 v16bf;
typedef __attribute__((ext_vector_type(8)))  float  v8f;
typedef __attribute__((ext_vector_type(4)))  unsigned int v4u;
typedef __attribute__((ext_vector_type(8)))  int v8i;
typedef __attribute__((ext_vector_type(4)))  int v4i;

union BfFrag {
    unsigned short u[16];
    v16bf v;
};

__device__ __forceinline__ unsigned short f2bf(float f) {
    // round-to-nearest-even fp32 -> bf16
    unsigned int u = __float_as_uint(f);
    unsigned int r = u + 0x7FFFu + ((u >> 16) & 1u);
    return (unsigned short)(r >> 16);
}

// A fragment: 16x32 bf16 tile (M x K), rows row-major in LDS with stride DFEAT.
// ISA 7.12.2: lanes 0-15 hold M=lane; VGPR i<4 -> K=2i,2i+1 (+half*8), VGPR 4-7 -> +16.
__device__ __forceinline__ v16bf load_a_frag(const unsigned short* __restrict__ lds,
                                             int lane, int k0) {
    int m = lane & 15, half = lane >> 4;
    BfFrag f;
#pragma unroll
    for (int i = 0; i < 8; ++i) {
        int kk = k0 + ((i & 4) ? 16 : 0) + half * 8 + (i & 3) * 2;
        f.u[2 * i]     = lds[m * DFEAT + kk];
        f.u[2 * i + 1] = lds[m * DFEAT + kk + 1];
    }
    return f.v;
}

// B fragment from pre-swizzled weights: lane's 16 bf16 values are contiguous.
// Layout index: (((nt*4 + kt)*32 + lane)*16 + j), 32-byte aligned per lane.
__device__ __forceinline__ v16bf load_b_packed(const unsigned short* __restrict__ bp,
                                               int nt, int kt, int lane) {
    return *(const v16bf*)(bp + (((nt * 4 + kt) * 32 + lane) << 4));
}

__device__ __forceinline__ v8f wmma_bf16(v16bf a, v16bf b, v8f c) {
    // (neg_a, A, neg_b, B, c_mod, C, reuse_a, reuse_b)
    return __builtin_amdgcn_wmma_f32_16x16x32_bf16(false, a, false, b, (short)0, c,
                                                   false, false);
}

// TDM: stage a (rows x 128) bf16 tile from global into LDS. Wave-level op.
__device__ __forceinline__ void tdm_load_tile_bf16(const unsigned short* gsrc,
                                                   unsigned int lds_off, int rows) {
    unsigned long long ga = (unsigned long long)(uintptr_t)gsrc;
    // D# group 0: count=1 | lds_addr | global_addr[56:0] | type=2 (bits 127:126)
    v4u g0;
    g0[0] = 1u;                         // count=1, user mode, no gather
    g0[1] = lds_off;                    // lds_addr
    g0[2] = (unsigned int)ga;           // global_addr[31:0]
    g0[3] = ((unsigned int)(ga >> 32) & 0x01FFFFFFu) | 0x80000000u;  // addr[56:32]|type=2
    // D# group 1: data_size=2B (code 1); tensor 128 x rows; tile 128 x 16; stride 128
    v8i g1;
    g1[0] = 1 << 16;                    // workgroup_mask=0, data_size=1 (2 bytes)
    g1[1] = DFEAT << 16;                // tensor_dim0[15:0] @ bits 63:48
    g1[2] = (rows & 0xFFFF) << 16;      // tensor_dim0[31:16]=0 | tensor_dim1[15:0]
    g1[3] = DFEAT << 16;                // tensor_dim1[31:16]=0 | tile_dim0=128
    g1[4] = 16;                         // tile_dim1=16 | tile_dim2=0
    g1[5] = DFEAT;                      // tensor_dim0_stride[31:0] = 128
    g1[6] = 0;                          // stride hi | tensor_dim1_stride lo
    g1[7] = 0;
    v4i gz4 = {0, 0, 0, 0};
    v8i gz8 = {0, 0, 0, 0, 0, 0, 0, 0};
    // 6-arg toolchain form: (g0, g1, g2, g3, g4, cpol)
    __builtin_amdgcn_tensor_load_to_lds(g0, g1, gz4, gz4, gz8, 0);
    __builtin_amdgcn_s_wait_tensorcnt(0);
}

// ---------------------------------------------------------------------------
// K0: (a) pre-swizzle the 4 weight matrices into bf16 WMMA B-fragment layout,
//     (b) convert x -> bf16, (c) zero the scatter accumulator.
// ---------------------------------------------------------------------------
__global__ void __launch_bounds__(256)
prep_kernel(const float* __restrict__ x,
            const float* __restrict__ Wq, const float* __restrict__ Wk,
            const float* __restrict__ Wv, const float* __restrict__ Wo,
            unsigned short* __restrict__ WqP, unsigned short* __restrict__ WkP,
            unsigned short* __restrict__ WvP, unsigned short* __restrict__ WoP,
            unsigned short* __restrict__ xbf, float* __restrict__ agg, int nd) {
    int tid = blockIdx.x * 256 + threadIdx.x;
    if (tid < 4 * DFEAT * DFEAT) {
        int m = tid >> 14;                     // which matrix
        int r = tid & (DFEAT * DFEAT - 1);     // fragment-layout index
        int j    = r & 15;
        int lane = (r >> 4) & 31;
        int kt   = (r >> 9) & 3;
        int nt   = (r >> 11) & 7;
        int i    = j >> 1;
        int half = lane >> 4;
        int col  = nt * 16 + (lane & 15);      // output feature n
        int kk   = kt * 32 + ((i & 4) ? 16 : 0) + half * 8 + (i & 3) * 2 + (j & 1);
        const float* W = (m == 0) ? Wq : (m == 1) ? Wk : (m == 2) ? Wv : Wo;
        unsigned short* P = (m == 0) ? WqP : (m == 1) ? WkP : (m == 2) ? WvP : WoP;
        // B[k][n] = W[n][k]  (torch Linear: y = x W^T)
        P[r] = f2bf(W[col * DFEAT + kk]);
    }
    if (tid < nd) {
        xbf[tid] = f2bf(x[tid]);
        agg[tid] = 0.0f;
    }
}

// ---------------------------------------------------------------------------
// K1: Q = x Wq^T, K = x Wk^T, V = x Wv^T.  One block = 16 rows.
// x tile staged into LDS by the Tensor Data Mover; 12 WMMAs per wave.
// ---------------------------------------------------------------------------
__global__ void __launch_bounds__(256)
qkv_gemm(const unsigned short* __restrict__ xbf,
         const unsigned short* __restrict__ WqP,
         const unsigned short* __restrict__ WkP,
         const unsigned short* __restrict__ WvP,
         float* __restrict__ Q, float* __restrict__ Kp, float* __restrict__ Vp,
         int N) {
    __shared__ __align__(16) unsigned short xs[16 * DFEAT];
    int row0 = blockIdx.x * 16;
    bool fullTile = (row0 + 16 <= N);

    if (threadIdx.x < 32) {
        int rows = N - row0;
        if (rows > 16) rows = 16;
        tdm_load_tile_bf16(xbf + (size_t)row0 * DFEAT,
                           (unsigned int)(uintptr_t)&xs[0], rows);
    }
    __syncthreads();

    int wave = threadIdx.x >> 5, lane = threadIdx.x & 31;

    v16bf a0 = load_a_frag(xs, lane, 0);
    v16bf a1 = load_a_frag(xs, lane, 32);
    v16bf a2 = load_a_frag(xs, lane, 64);
    v16bf a3 = load_a_frag(xs, lane, 96);

    const unsigned short* Ws[3] = {WqP, WkP, WvP};
    float* Os[3] = {Q, Kp, Vp};

    int col = wave * 16 + (lane & 15), half = lane >> 4;

#pragma unroll
    for (int m = 0; m < 3; ++m) {
        v8f c = {0.f, 0.f, 0.f, 0.f, 0.f, 0.f, 0.f, 0.f};
        c = wmma_bf16(a0, load_b_packed(Ws[m], wave, 0, lane), c);
        c = wmma_bf16(a1, load_b_packed(Ws[m], wave, 1, lane), c);
        c = wmma_bf16(a2, load_b_packed(Ws[m], wave, 2, lane), c);
        c = wmma_bf16(a3, load_b_packed(Ws[m], wave, 3, lane), c);

        float* obase = Os[m] + (size_t)(row0 + half * 8) * DFEAT + col;
        if (fullTile) {
            // block-uniform fast path: no per-element exec-mask churn
#pragma unroll
            for (int r = 0; r < 8; ++r) obase[(size_t)r * DFEAT] = c[r];
        } else {
#pragma unroll
            for (int r = 0; r < 8; ++r) {
                if (row0 + half * 8 + r < N) obase[(size_t)r * DFEAT] = c[r];
            }
        }
    }
}

// ---------------------------------------------------------------------------
// K2: per-edge gated message + scatter-add.  One wave32 per edge.
// Gathers/atomics hit L2 (Q/K/V/agg = 102 MB < 192 MB L2).
// ---------------------------------------------------------------------------
__global__ void __launch_bounds__(256)
edge_attend(const int* __restrict__ src, const int* __restrict__ dst,
            const float* __restrict__ Q, const float* __restrict__ Kp,
            const float* __restrict__ Vp, float* __restrict__ agg, int E) {
    int wave = threadIdx.x >> 5, lane = threadIdx.x & 31;
    int e = blockIdx.x * 8 + wave;
    if (e >= E) return;

    int s = src[e], d = dst[e];
    float4 qv = ((const float4*)(Q  + (size_t)d * DFEAT))[lane];
    float4 kv = ((const float4*)(Kp + (size_t)s * DFEAT))[lane];
    float p = qv.x * kv.x + qv.y * kv.y + qv.z * kv.z + qv.w * kv.w;
#pragma unroll
    for (int off = 16; off > 0; off >>= 1) p += __shfl_xor(p, off, 32);

    float logit = p * 0.08838834764831845f;  // 1/sqrt(128)
    float g = 1.0f / (1.0f + __expf(-logit));

    float4 vv = ((const float4*)(Vp + (size_t)s * DFEAT))[lane];
    float* ap = agg + (size_t)d * DFEAT + lane * 4;
    atomicAdd(ap + 0, g * vv.x);
    atomicAdd(ap + 1, g * vv.y);
    atomicAdd(ap + 2, g * vv.z);
    atomicAdd(ap + 3, g * vv.w);
}

// ---------------------------------------------------------------------------
// K3: out = agg Wo^T + bo; h = x + out; LayerNorm(h).  One block = 16 rows.
// ---------------------------------------------------------------------------
__global__ void __launch_bounds__(256)
out_ln(const float* __restrict__ agg, const unsigned short* __restrict__ WoP,
       const float* __restrict__ x, const float* __restrict__ bo,
       const float* __restrict__ gamma, const float* __restrict__ beta,
       float* __restrict__ out, int N) {
    __shared__ __align__(16) unsigned short as_[16 * DFEAT];
    __shared__ __align__(16) float hs[16 * DFEAT];
    int row0 = blockIdx.x * 16;
    bool fullTile = (row0 + 16 <= N);

    if (fullTile) {
        for (int idx = threadIdx.x; idx < 16 * DFEAT; idx += 256)
            as_[idx] = f2bf(agg[(size_t)row0 * DFEAT + idx]);
    } else {
        for (int idx = threadIdx.x; idx < 16 * DFEAT; idx += 256) {
            int gr = row0 + (idx >> 7);
            as_[idx] = f2bf((gr < N) ? agg[(size_t)gr * DFEAT + (idx & (DFEAT - 1))]
                                     : 0.0f);
        }
    }
    __syncthreads();

    int wave = threadIdx.x >> 5, lane = threadIdx.x & 31;

    v8f c = {0.f, 0.f, 0.f, 0.f, 0.f, 0.f, 0.f, 0.f};
#pragma unroll
    for (int kt = 0; kt < 4; ++kt) {
        v16bf a = load_a_frag(as_, lane, kt * 32);
        v16bf b = load_b_packed(WoP, wave, kt, lane);
        c = wmma_bf16(a, b, c);
    }
    int col = wave * 16 + (lane & 15), half = lane >> 4;
#pragma unroll
    for (int r = 0; r < 8; ++r) hs[(half * 8 + r) * DFEAT + col] = c[r];
    __syncthreads();

    // LayerNorm + residual: wave w handles rows w and w+8 of the tile.
#pragma unroll
    for (int rep = 0; rep < 2; ++rep) {
        int r = wave + rep * 8;
        int gr = row0 + r;
        if (gr >= N) continue;

        float4 xv = ((const float4*)(x + (size_t)gr * DFEAT))[lane];
        float4 bv = ((const float4*)bo)[lane];
        float4 ov = ((const float4*)(hs + r * DFEAT))[lane];
        float4 hv;
        hv.x = xv.x + ov.x + bv.x;
        hv.y = xv.y + ov.y + bv.y;
        hv.z = xv.z + ov.z + bv.z;
        hv.w = xv.w + ov.w + bv.w;

        float sum = hv.x + hv.y + hv.z + hv.w;
#pragma unroll
        for (int off = 16; off > 0; off >>= 1) sum += __shfl_xor(sum, off, 32);
        float mu = sum * (1.0f / DFEAT);

        float dx0 = hv.x - mu, dx1 = hv.y - mu, dx2 = hv.z - mu, dx3 = hv.w - mu;
        float vsum = dx0 * dx0 + dx1 * dx1 + dx2 * dx2 + dx3 * dx3;
#pragma unroll
        for (int off = 16; off > 0; off >>= 1) vsum += __shfl_xor(vsum, off, 32);
        float inv = rsqrtf(vsum * (1.0f / DFEAT) + LN_EPS);

        float4 gv = ((const float4*)gamma)[lane];
        float4 bt = ((const float4*)beta)[lane];
        float4 o;
        o.x = dx0 * inv * gv.x + bt.x;
        o.y = dx1 * inv * gv.y + bt.y;
        o.z = dx2 * inv * gv.z + bt.z;
        o.w = dx3 * inv * gv.w + bt.w;
        ((float4*)(out + (size_t)gr * DFEAT))[lane] = o;
    }
}

// ---------------------------------------------------------------------------
extern "C" void kernel_launch(void* const* d_in, const int* in_sizes, int n_in,
                              void* d_out, int out_size, void* d_ws, size_t ws_size,
                              hipStream_t stream) {
    const float* x     = (const float*)d_in[0];
    const int*   edges = (const int*)d_in[1];
    const float* Wq    = (const float*)d_in[2];
    const float* Wk    = (const float*)d_in[3];
    const float* Wv    = (const float*)d_in[4];
    const float* Wo    = (const float*)d_in[5];
    const float* bo    = (const float*)d_in[6];
    const float* gamma = (const float*)d_in[7];
    const float* beta  = (const float*)d_in[8];

    int N = in_sizes[0] / DFEAT;
    int E = in_sizes[1] / 2;
    const int* src = edges;
    const int* dst = edges + E;

    // Workspace: Q,K,V,agg fp32; xbf bf16; 4 pre-swizzled bf16 weight tables.
    char* ws = (char*)d_ws;
    size_t nd = (size_t)N * DFEAT;
    float* Q   = (float*)ws; ws += nd * sizeof(float);
    float* Kp  = (float*)ws; ws += nd * sizeof(float);
    float* Vp  = (float*)ws; ws += nd * sizeof(float);
    float* agg = (float*)ws; ws += nd * sizeof(float);
    unsigned short* xbf = (unsigned short*)ws; ws += nd * sizeof(unsigned short);
    unsigned short* WqP = (unsigned short*)ws; ws += DFEAT * DFEAT * 2;
    unsigned short* WkP = (unsigned short*)ws; ws += DFEAT * DFEAT * 2;
    unsigned short* WvP = (unsigned short*)ws; ws += DFEAT * DFEAT * 2;
    unsigned short* WoP = (unsigned short*)ws; ws += DFEAT * DFEAT * 2;

    int prepN = (int)nd > 4 * DFEAT * DFEAT ? (int)nd : 4 * DFEAT * DFEAT;
    prep_kernel<<<(prepN + 255) / 256, 256, 0, stream>>>(
        x, Wq, Wk, Wv, Wo, WqP, WkP, WvP, WoP, xbf, agg, (int)nd);

    int nTiles = (N + 15) / 16;
    qkv_gemm<<<nTiles, 256, 0, stream>>>(xbf, WqP, WkP, WvP, Q, Kp, Vp, N);

    edge_attend<<<(E + 7) / 8, 256, 0, stream>>>(src, dst, Q, Kp, Vp, agg, E);

    out_ln<<<nTiles, 256, 0, stream>>>(agg, WoP, x, bo, gamma, beta,
                                       (float*)d_out, N);
}
